// GCM_64879775973997
// MI455X (gfx1250) — matrix operand order repaired and checked by reading, not compiled, not run on hardware
//
#include <hip/hip_runtime.h>
#include <hip/hip_bf16.h>

#define NUM_USERS        50000
#define NUM_ITEMS        22000
#define NUM_VALID_ITEMS  20000
#define NUM_CONTEXTS     100000
#define NUM_INSTS        500000
#define HIDDEN           64
#define BATCH            4096

typedef __attribute__((ext_vector_type(16))) _Float16 v16h;
typedef __attribute__((ext_vector_type(8)))  _Float16 v8h;
typedef __attribute__((ext_vector_type(8)))  float    v8f;

// ---------------------------------------------------------------------------
// zero workspace buffer
__global__ void zero_kernel(float* __restrict__ p, int n) {
  int i = blockIdx.x * blockDim.x + threadIdx.x;
  if (i < n) p[i] = 0.0f;
}

// ---------------------------------------------------------------------------
// encoded = (base_row + sum of 3 gathered feat rows) / 4
// layer = encoded ; fin = encoded * (1/3)      (wave32: one row per wave)
__global__ void encode_kernel(const float* __restrict__ base,
                              const float* __restrict__ feat,
                              const int*   __restrict__ mat,
                              float* __restrict__ layer,
                              float* __restrict__ fin, int N) {
  int w    = (blockIdx.x * blockDim.x + threadIdx.x) >> 5;
  int lane = threadIdx.x & 31;
  if (w >= N) return;
  int d = lane * 2;
  const int* row = mat + (size_t)w * 3;
  const float* b = base + (size_t)w * HIDDEN + d;
  float s0 = b[0], s1 = b[1];
#pragma unroll
  for (int f = 0; f < 3; ++f) {
    const float* fr = feat + (size_t)row[f] * HIDDEN + d;
    s0 += fr[0]; s1 += fr[1];
  }
  s0 *= 0.25f; s1 *= 0.25f;
  size_t o = (size_t)w * HIDDEN + d;
  layer[o] = s0; layer[o + 1] = s1;
  const float W = 1.0f / 3.0f;
  fin[o] = s0 * W; fin[o + 1] = s1 * W;
}

// ---------------------------------------------------------------------------
// ctx_mean[c] = (c_feat[cm0] + c_feat[cm1] + c_feat[cm2] + item_emb[cm3]) / 4
__global__ void ctx_mean_kernel(const float* __restrict__ c_feat,
                                const float* __restrict__ item_emb,
                                const int*   __restrict__ cmat,
                                float* __restrict__ out, int N) {
  int w    = (blockIdx.x * blockDim.x + threadIdx.x) >> 5;
  int lane = threadIdx.x & 31;
  if (w >= N) return;
  int d = lane * 2;
  const int* row = cmat + (size_t)w * 4;
  float s0 = 0.0f, s1 = 0.0f;
#pragma unroll
  for (int f = 0; f < 3; ++f) {
    const float* fr = c_feat + (size_t)row[f] * HIDDEN + d;
    s0 += fr[0]; s1 += fr[1];
  }
  const float* ir = item_emb + (size_t)row[3] * HIDDEN + d;
  s0 += ir[0]; s1 += ir[1];
  size_t o = (size_t)w * HIDDEN + d;
  out[o] = s0 * 0.25f; out[o + 1] = s1 * 0.25f;
}

// ---------------------------------------------------------------------------
// per-edge: next_u[uid] += layer_i[iid] + ctx ; next_i[iid] += layer_u[uid] + ctx
// one wave per edge, float2 per lane; scatter via f32 atomics (resolve in L2)
__global__ void gcn_edge_kernel(const float* __restrict__ layer_u,
                                const float* __restrict__ layer_i,
                                const float* __restrict__ ctxm,
                                const int*   __restrict__ uid,
                                const int*   __restrict__ iid,
                                const int*   __restrict__ cid,
                                float* __restrict__ next_u,
                                float* __restrict__ next_i) {
  int e    = (blockIdx.x * blockDim.x + threadIdx.x) >> 5;
  int lane = threadIdx.x & 31;
  if (e >= NUM_INSTS) return;
  // stream the index arrays ahead (global_prefetch_b8)
  if (e + 4096 < NUM_INSTS) {
    __builtin_prefetch(&uid[e + 4096], 0, 0);
    __builtin_prefetch(&iid[e + 4096], 0, 0);
    __builtin_prefetch(&cid[e + 4096], 0, 0);
  }
  int u = uid[e], it = iid[e], c = cid[e];
  int d = lane * 2;
  const float* cr = ctxm    + (size_t)c  * HIDDEN + d;
  const float* ur = layer_u + (size_t)u  * HIDDEN + d;
  const float* ir = layer_i + (size_t)it * HIDDEN + d;
  float c0 = cr[0], c1 = cr[1];
  float u0 = ur[0], u1 = ur[1];
  float i0 = ir[0], i1 = ir[1];
  float* pu = next_u + (size_t)u  * HIDDEN + d;
  float* pi = next_i + (size_t)it * HIDDEN + d;
  atomicAdd(pu,     i0 + c0);
  atomicAdd(pu + 1, i1 + c1);
  atomicAdd(pi,     u0 + c0);
  atomicAdd(pi + 1, u1 + c1);
}

// ---------------------------------------------------------------------------
// fin += next * (1/3)
__global__ void accum_kernel(float* __restrict__ fin,
                             const float* __restrict__ next, int n) {
  int i = blockIdx.x * blockDim.x + threadIdx.x;
  if (i < n) fin[i] += next[i] * (1.0f / 3.0f);
}

// ---------------------------------------------------------------------------
__device__ __forceinline__ void store_row_f16(_Float16* dst, const float* src, int d) {
  union { _Float16 h[2]; unsigned int u; } pk;
  pk.h[0] = (_Float16)src[0];
  pk.h[1] = (_Float16)src[1];
  *(unsigned int*)(dst + d) = pk.u;
}

// FM decoder: one wave per batch element. X = 16x64 f16 in LDS (rows 0..5 live,
// 6..15 zero). G = X*X^T via two v_wmma_f32_16x16x32_f16 (K=0..31, 32..63).
// interaction = 0.5 * (sum(G[0:6,0:6]) - trace) = sum_{f<g} <x_f, x_g>.
__global__ void __launch_bounds__(256)
fm_kernel(const int* __restrict__ uid, const int* __restrict__ iid,
          const int* __restrict__ cidx, const int* __restrict__ cmat,
          const float* __restrict__ c_feat, const float* __restrict__ item_emb,
          const float* __restrict__ fin_u, const float* __restrict__ fin_i,
          const float* __restrict__ user_bias, const float* __restrict__ item_bias,
          const float* __restrict__ gbias, float* __restrict__ out) {
  __shared__ __align__(32) _Float16 Xs[8][16][HIDDEN];  // 16 KB: 8 waves/block
  int w    = threadIdx.x >> 5;
  int lane = threadIdx.x & 31;
  int b    = blockIdx.x * 8 + w;   // BATCH == 512*8 exactly: no masking, EXEC all-1s
  int u  = uid[b];
  int it = iid[b];
  const int* cm = cmat + (size_t)cidx[b] * 4;
  int d = lane * 2;

  _Float16* X = &Xs[w][0][0];
  store_row_f16(X + 0 * HIDDEN, fin_u    + (size_t)u     * HIDDEN + d, d);
  store_row_f16(X + 1 * HIDDEN, fin_i    + (size_t)it    * HIDDEN + d, d);
  store_row_f16(X + 2 * HIDDEN, c_feat   + (size_t)cm[0] * HIDDEN + d, d);
  store_row_f16(X + 3 * HIDDEN, c_feat   + (size_t)cm[1] * HIDDEN + d, d);
  store_row_f16(X + 4 * HIDDEN, c_feat   + (size_t)cm[2] * HIDDEN + d, d);
  store_row_f16(X + 5 * HIDDEN, item_emb + (size_t)cm[3] * HIDDEN + d, d);
#pragma unroll
  for (int r = 6; r < 16; ++r)
    *(unsigned int*)(X + r * HIDDEN + d) = 0u;
  __syncthreads();

  // Fragment layouts used here, contiguous k, not ISA table order (16-bit A 16x32; B 32x16):
  //   A: lanes 0-15 -> row M=lane,   K=0..15
  //      lanes 16-31-> row M=lane-16, K=16..31
  //   B: lanes 0-15 -> col N=lane,    K=0..15 ; lanes 16-31 -> col N, K=16..31
  int M     = lane & 15;
  int kbase = (lane < 16) ? 0 : 8;
  int koff  = (lane < 16) ? 0 : 16;

  v8f acc = {};
#pragma unroll
  for (int kk = 0; kk < 2; ++kk) {
    int K0 = kk * 32;
    v8h lo = *(const v8h*)(X + M * HIDDEN + K0 + koff);        // K0+koff .. +7
    v8h hi = *(const v8h*)(X + M * HIDDEN + K0 + koff + 8);    // K0+koff+8 .. +15
    v16h a;
#pragma unroll
    for (int i = 0; i < 8; ++i) { a[i] = lo[i]; a[8 + i] = hi[i]; }
    // B[k][n] = X[n][k]; column N == lane&15 == M here
    v16h bb = *(const v16h*)(X + M * HIDDEN + K0 + koff);
    acc = __builtin_amdgcn_wmma_f32_16x16x32_f16(
        /*neg_a=*/false, a, /*neg_b=*/false, bb,
        /*c_mod=*/(short)0, acc, /*reuse_a=*/false, /*reuse_b=*/false);
  }

  // D layout: acc[r] holds G[r + 8*(lane/16)][lane%16].
  // Live 6x6 block => lanes 0..5 (rows 0..5 in acc[0..5], col N=lane).
  float s = 0.0f;
  if (lane < 6)
    s = acc[0] + acc[1] + acc[2] + acc[3] + acc[4] + acc[5] - acc[lane];
#pragma unroll
  for (int off = 16; off > 0; off >>= 1) s += __shfl_down(s, off, 32);
  if (lane == 0)
    out[b] = 0.5f * s + user_bias[u] + item_bias[it] + gbias[0];
}

// ---------------------------------------------------------------------------
extern "C" void kernel_launch(void* const* d_in, const int* in_sizes, int n_in,
                              void* d_out, int out_size, void* d_ws, size_t ws_size,
                              hipStream_t stream) {
  const int*   user_id    = (const int*)d_in[0];
  const int*   item_id    = (const int*)d_in[1];
  const int*   context_id = (const int*)d_in[2];
  const float* user_emb   = (const float*)d_in[3];
  const float* item_emb   = (const float*)d_in[4];
  const float* u_feat     = (const float*)d_in[5];
  const float* i_feat     = (const float*)d_in[6];
  const float* c_feat     = (const float*)d_in[7];
  const float* user_bias  = (const float*)d_in[8];
  const float* item_bias  = (const float*)d_in[9];
  const float* gbias      = (const float*)d_in[10];
  const int*   umat       = (const int*)d_in[11];
  const int*   imat       = (const int*)d_in[12];
  const int*   cmat       = (const int*)d_in[13];
  const int*   e_uid      = (const int*)d_in[14];
  const int*   e_iid      = (const int*)d_in[15];
  const int*   e_cid      = (const int*)d_in[16];

  const size_t UN = (size_t)NUM_USERS * HIDDEN;        // 3.2M floats
  const size_t IN = (size_t)NUM_VALID_ITEMS * HIDDEN;  // 1.28M floats
  const size_t CN = (size_t)NUM_CONTEXTS * HIDDEN;     // 6.4M floats
  float* ws      = (float*)d_ws;                        // needs ~80 MB total
  float* layer_u = ws;
  float* next_u  = layer_u + UN;
  float* layer_i = next_u + UN;
  float* next_i  = layer_i + IN;
  float* fin_u   = next_i + IN;
  float* fin_i   = fin_u + UN;
  float* ctxm    = fin_i + IN;
  (void)CN; (void)in_sizes; (void)n_in; (void)out_size; (void)ws_size;

  dim3 blk(256);
  encode_kernel<<<(NUM_USERS + 7) / 8, blk, 0, stream>>>(
      user_emb, u_feat, umat, layer_u, fin_u, NUM_USERS);
  encode_kernel<<<(NUM_VALID_ITEMS + 7) / 8, blk, 0, stream>>>(
      item_emb, i_feat, imat, layer_i, fin_i, NUM_VALID_ITEMS);
  ctx_mean_kernel<<<(NUM_CONTEXTS + 7) / 8, blk, 0, stream>>>(
      c_feat, item_emb, cmat, ctxm, NUM_CONTEXTS);

  float* cu = layer_u; float* ci = layer_i;
  float* nu = next_u;  float* ni = next_i;
  for (int k = 0; k < 2; ++k) {
    zero_kernel<<<(int)((UN + 255) / 256), blk, 0, stream>>>(nu, (int)UN);
    zero_kernel<<<(int)((IN + 255) / 256), blk, 0, stream>>>(ni, (int)IN);
    gcn_edge_kernel<<<(NUM_INSTS + 7) / 8, blk, 0, stream>>>(
        cu, ci, ctxm, e_uid, e_iid, e_cid, nu, ni);
    accum_kernel<<<(int)((UN + 255) / 256), blk, 0, stream>>>(fin_u, nu, (int)UN);
    accum_kernel<<<(int)((IN + 255) / 256), blk, 0, stream>>>(fin_i, ni, (int)IN);
    float* t;
    t = cu; cu = nu; nu = t;
    t = ci; ci = ni; ni = t;
  }

  fm_kernel<<<BATCH / 8, blk, 0, stream>>>(
      user_id, item_id, context_id, cmat, c_feat, item_emb,
      fin_u, fin_i, user_bias, item_bias, gbias, (float*)d_out);
}
